// SelfAttention_2757369004240
// MI455X (gfx1250) — matrix operand order repaired and checked by reading, not compiled, not run on hardware
//
#include <hip/hip_runtime.h>

#define S_LEN   4096
#define DMODEL  768
#define NHEADS  12
#define DHEAD   64
#define QKVN    2304

typedef __bf16 bf16;
typedef __attribute__((ext_vector_type(8)))  __bf16 v8bf;
typedef __attribute__((ext_vector_type(16))) __bf16 v16bf;
typedef __attribute__((ext_vector_type(8)))  float  v8f;

__device__ __forceinline__ v8f wmma_bf16(v16bf a, v16bf b, v8f c) {
    return __builtin_amdgcn_wmma_f32_16x16x32_bf16(
        /*neg_a=*/false, a, /*neg_b=*/false, b,
        /*c_mod=*/(short)0, c, /*reuse_a=*/false, /*reuse_b=*/false);
}

// A operand: 16x32 (MxK) tile from row-major [M][K] memory.
// Lane (kg=lane>>4, r=lane&15): row r, K = {kg*8..+7, 16+kg*8..+7}
__device__ __forceinline__ v16bf load_a_frag(const bf16* p, int ld) {
    const int lane = threadIdx.x & 31;
    const bf16* row = p + (size_t)(lane & 15) * ld;
    const int ko = (lane >> 4) * 8;
    v8bf lo = *(const v8bf*)(row + ko);
    v8bf hi = *(const v8bf*)(row + 16 + ko);
    return __builtin_shufflevector(lo, hi, 0,1,2,3,4,5,6,7,8,9,10,11,12,13,14,15);
}

// B operand: 32x16 (KxN) tile from row-major [N][K] memory (GEMM = A @ Mem^T).
// Lane: row n=lane&15, K = {kg*8..+7, 16+kg*8..+7}
__device__ __forceinline__ v16bf load_b_frag(const bf16* p, int ld) {
    const int lane = threadIdx.x & 31;
    const bf16* row = p + (size_t)(lane & 15) * ld;
    const int ko = (lane >> 4) * 8;
    v8bf lo = *(const v8bf*)(row + ko);
    v8bf hi = *(const v8bf*)(row + 16 + ko);
    return __builtin_shufflevector(lo, hi, 0,1,2,3,4,5,6,7,8,9,10,11,12,13,14,15);
}

// ---------------------------------------------------------------- convert (8-wide)
__global__ void cvt_f32_bf16(const float* __restrict__ in,
                             bf16* __restrict__ out, int n) {
    int i = (blockIdx.x * blockDim.x + threadIdx.x) * 8;
    if (i < n) {
        v8f v = *(const v8f*)(in + i);
        v8bf o;
#pragma unroll
        for (int k = 0; k < 8; ++k) o[k] = (bf16)v[k];
        *(v8bf*)(out + i) = o;
    }
}

// ---------------------------------------------------------------- QKV GEMM (transposed)
// Computes qkv^T[n][m] = w_in[n][:].x[m][:] + b_in[n]; lane elements run along the
// feature dim so Q/K writes are packed b128 stores. Q pre-scaled by 1/sqrt(dh).
__global__ __launch_bounds__(256)
void qkv_gemm(const bf16* __restrict__ xb, const bf16* __restrict__ wb,
              const float* __restrict__ bias,
              bf16* __restrict__ qh, bf16* __restrict__ kh, bf16* __restrict__ vt) {
    const int wave = threadIdx.x >> 5;
    const int lane = threadIdx.x & 31;
    const int r = lane & 15, kg = lane >> 4;
    const int fbase = blockIdx.x * 128 + (wave & 3) * 32;   // feature (0..2304)
    const int tbase = blockIdx.y * 64 + (wave >> 2) * 32;   // token   (0..4096)

    v8f acc[2][2] = {};
#pragma unroll 4
    for (int k0 = 0; k0 < DMODEL; k0 += 32) {
        v16bf a0 = load_a_frag(wb + (size_t)fbase * DMODEL + k0, DMODEL);
        v16bf a1 = load_a_frag(wb + (size_t)(fbase + 16) * DMODEL + k0, DMODEL);
        v16bf b0 = load_b_frag(xb + (size_t)tbase * DMODEL + k0, DMODEL);
        v16bf b1 = load_b_frag(xb + (size_t)(tbase + 16) * DMODEL + k0, DMODEL);
        acc[0][0] = wmma_bf16(a0, b0, acc[0][0]);
        acc[0][1] = wmma_bf16(a0, b1, acc[0][1]);
        acc[1][0] = wmma_bf16(a1, b0, acc[1][0]);
        acc[1][1] = wmma_bf16(a1, b1, acc[1][1]);
    }

#pragma unroll
    for (int mi = 0; mi < 2; ++mi) {
        const int nrow = fbase + mi * 16 + 8 * kg;       // first of 8 consecutive features
        const v8f bv = *(const v8f*)(bias + nrow);
#pragma unroll
        for (int ni = 0; ni < 2; ++ni) {
            const int m = tbase + ni * 16 + r;           // token
            if (nrow < DMODEL) {                         // ---- Q (scaled)
                int h = nrow >> 6, d0 = nrow & 63;
                v8bf pk;
#pragma unroll
                for (int i = 0; i < 8; ++i) pk[i] = (bf16)((acc[mi][ni][i] + bv[i]) * 0.125f);
                *(v8bf*)(qh + ((size_t)h * S_LEN + m) * DHEAD + d0) = pk;
            } else if (nrow < 2 * DMODEL) {              // ---- K
                int nn = nrow - DMODEL;
                int h = nn >> 6, d0 = nn & 63;
                v8bf pk;
#pragma unroll
                for (int i = 0; i < 8; ++i) pk[i] = (bf16)(acc[mi][ni][i] + bv[i]);
                *(v8bf*)(kh + ((size_t)h * S_LEN + m) * DHEAD + d0) = pk;
            } else {                                     // ---- V, transposed [h][d][s]
                int nn = nrow - 2 * DMODEL;
                int h = nn >> 6, d0 = nn & 63;
#pragma unroll
                for (int i = 0; i < 8; ++i)
                    vt[((size_t)h * DHEAD + d0 + i) * S_LEN + m] = (bf16)(acc[mi][ni][i] + bv[i]);
            }
        }
    }
}

// ---------------------------------------------------------------- attention
// One wave owns 32 query columns; scores computed TRANSPOSED (S^T = K @ Q^T) so
// each lane owns whole query columns: softmax reductions are in-lane trees plus
// a single shfl_xor(16). O is accumulated transposed (O^T = V^T @ P^T).
__global__ __launch_bounds__(256)
void attention(const bf16* __restrict__ qh, const bf16* __restrict__ kh,
               const bf16* __restrict__ vt, bf16* __restrict__ ao,
               const int* __restrict__ causal_flag) {
    __shared__ bf16 Plds[8][32 * 32];                    // per-wave [q][k] tile

    const int wave = threadIdx.x >> 5;
    const int lane = threadIdx.x & 31;
    const int r = lane & 15, kg = lane >> 4;
    const int h = blockIdx.y;
    const int qbase = blockIdx.x * 256 + wave * 32;
    const int causal = *causal_flag;

    const bf16* Q = qh + (size_t)h * S_LEN * DHEAD;
    const bf16* K = kh + (size_t)h * S_LEN * DHEAD;
    const bf16* V = vt + (size_t)h * DHEAD * S_LEN;

    // resident Q fragments as B operands: [q-tile][k-step]
    v16bf qB[2][2];
#pragma unroll
    for (int nt = 0; nt < 2; ++nt)
#pragma unroll
        for (int ks = 0; ks < 2; ++ks)
            qB[nt][ks] = load_b_frag(Q + (size_t)(qbase + nt * 16) * DHEAD + ks * 32, DHEAD);

    v8f O[4][2] = {};                                    // O^T: [d-tile][q-tile]
    float M[2] = { -1e30f, -1e30f }, L[2] = { 0.f, 0.f };

    bf16* P = Plds[wave];
    const int jend = causal ? qbase : (S_LEN - 32);

    for (int j = 0; j <= jend; j += 32) {
        // ---- S^T[k][q] = K @ Q^T
        v8f s[2][2] = {};
#pragma unroll
        for (int ks = 0; ks < 2; ++ks)
#pragma unroll
            for (int mt = 0; mt < 2; ++mt) {
                v16bf ka = load_a_frag(K + (size_t)(j + mt * 16) * DHEAD + ks * 32, DHEAD);
                s[mt][0] = wmma_bf16(ka, qB[0][ks], s[mt][0]);
                s[mt][1] = wmma_bf16(ka, qB[1][ks], s[mt][1]);
            }

        if (causal && (j == qbase)) {                    // diagonal block mask
#pragma unroll
            for (int mt = 0; mt < 2; ++mt)
#pragma unroll
                for (int nt = 0; nt < 2; ++nt)
#pragma unroll
                    for (int i = 0; i < 8; ++i) {
                        int kglob = j + mt * 16 + 8 * kg + i;
                        int qglob = qbase + nt * 16 + r;
                        if (kglob > qglob) s[mt][nt][i] = -1e30f;
                    }
        }

        // ---- online softmax: each lane owns query column (nt*16 + r)
#pragma unroll
        for (int nt = 0; nt < 2; ++nt) {
            float mx = s[0][nt][0];
#pragma unroll
            for (int i = 1; i < 8; ++i) mx = fmaxf(mx, s[0][nt][i]);
#pragma unroll
            for (int i = 0; i < 8; ++i) mx = fmaxf(mx, s[1][nt][i]);
            mx = fmaxf(mx, __shfl_xor(mx, 16, 32));      // combine key halves
            float Mnew = fmaxf(M[nt], mx);
            float alpha = __expf(M[nt] - Mnew);
            M[nt] = Mnew;

            float sum = 0.f;
#pragma unroll
            for (int mt = 0; mt < 2; ++mt) {
                v8bf pk;
#pragma unroll
                for (int i = 0; i < 8; ++i) {
                    float p = __expf(s[mt][nt][i] - Mnew);
                    sum += p;
                    pk[i] = (bf16)p;
                }
                // P^T stored as [q][k]: 8 consecutive k per lane -> one b128 store
                *(v8bf*)(P + (size_t)(nt * 16 + r) * 32 + mt * 16 + 8 * kg) = pk;
            }
            sum += __shfl_xor(sum, 16, 32);
            L[nt] = L[nt] * alpha + sum;
#pragma unroll
            for (int dt = 0; dt < 4; ++dt)
#pragma unroll
                for (int i = 0; i < 8; ++i) O[dt][nt][i] *= alpha;
        }

        // ---- O^T += V^T(A) @ P^T(B)
        v16bf pf0 = load_b_frag(P, 32);
        v16bf pf1 = load_b_frag(P + 16 * 32, 32);
#pragma unroll
        for (int dt = 0; dt < 4; ++dt) {
            v16bf va = load_a_frag(V + (size_t)(dt * 16) * S_LEN + j, S_LEN);
            O[dt][0] = wmma_bf16(va, pf0, O[dt][0]);
            O[dt][1] = wmma_bf16(va, pf1, O[dt][1]);
        }
    }

    // ---- normalize, emit ao[q][h*64+d] with packed stores
#pragma unroll
    for (int nt = 0; nt < 2; ++nt) {
        const float inv = 1.0f / L[nt];
        const int qglob = qbase + nt * 16 + r;
#pragma unroll
        for (int dt = 0; dt < 4; ++dt) {
            v8bf ov;
#pragma unroll
            for (int i = 0; i < 8; ++i) ov[i] = (bf16)(O[dt][nt][i] * inv);
            *(v8bf*)(ao + (size_t)qglob * DMODEL + h * DHEAD + dt * 16 + 8 * kg) = ov;
        }
    }
}

// ---------------------------------------------------------------- output proj (transposed)
// out[m][n] = ao[m][:].w_out[n][:] + b_out[n]; lane elements run along n so the
// epilogue is a pure vector bias-add + v8f store.
__global__ __launch_bounds__(256)
void out_proj(const bf16* __restrict__ ab, const bf16* __restrict__ wb,
              const float* __restrict__ bias, float* __restrict__ out) {
    const int wave = threadIdx.x >> 5;
    const int lane = threadIdx.x & 31;
    const int r = lane & 15, kg = lane >> 4;
    const int fbase = blockIdx.x * 128 + (wave & 3) * 32;   // feature (0..768)
    const int tbase = blockIdx.y * 64 + (wave >> 2) * 32;   // token   (0..4096)

    v8f acc[2][2] = {};
#pragma unroll 4
    for (int k0 = 0; k0 < DMODEL; k0 += 32) {
        v16bf a0 = load_a_frag(wb + (size_t)fbase * DMODEL + k0, DMODEL);
        v16bf a1 = load_a_frag(wb + (size_t)(fbase + 16) * DMODEL + k0, DMODEL);
        v16bf b0 = load_b_frag(ab + (size_t)tbase * DMODEL + k0, DMODEL);
        v16bf b1 = load_b_frag(ab + (size_t)(tbase + 16) * DMODEL + k0, DMODEL);
        acc[0][0] = wmma_bf16(a0, b0, acc[0][0]);
        acc[0][1] = wmma_bf16(a0, b1, acc[0][1]);
        acc[1][0] = wmma_bf16(a1, b0, acc[1][0]);
        acc[1][1] = wmma_bf16(a1, b1, acc[1][1]);
    }

#pragma unroll
    for (int mi = 0; mi < 2; ++mi) {
        const int nrow = fbase + mi * 16 + 8 * kg;
        const v8f bv = *(const v8f*)(bias + nrow);
#pragma unroll
        for (int ni = 0; ni < 2; ++ni) {
            const int m = tbase + ni * 16 + r;
            v8f v = acc[mi][ni];
#pragma unroll
            for (int i = 0; i < 8; ++i) v[i] += bv[i];
            *(v8f*)(out + (size_t)m * DMODEL + nrow) = v;
        }
    }
}

// ---------------------------------------------------------------- launch
extern "C" void kernel_launch(void* const* d_in, const int* in_sizes, int n_in,
                              void* d_out, int out_size, void* d_ws, size_t ws_size,
                              hipStream_t stream) {
    const float* x     = (const float*)d_in[0];
    const float* w_in  = (const float*)d_in[1];
    const float* b_in  = (const float*)d_in[2];
    const float* w_out = (const float*)d_in[3];
    const float* b_out = (const float*)d_in[4];
    const int*   flag  = (const int*)d_in[5];

    char* ws = (char*)d_ws;
    size_t off = 0;
    auto carve = [&](size_t elems) -> bf16* {
        off = (off + 255) & ~(size_t)255;
        bf16* p = (bf16*)(ws + off);
        off += elems * sizeof(unsigned short);
        return p;
    };
    bf16* xb  = carve((size_t)S_LEN * DMODEL);
    bf16* wib = carve((size_t)QKVN * DMODEL);
    bf16* wob = carve((size_t)DMODEL * DMODEL);
    bf16* qh  = carve((size_t)NHEADS * S_LEN * DHEAD);
    bf16* kh  = carve((size_t)NHEADS * S_LEN * DHEAD);
    bf16* vt  = carve((size_t)NHEADS * DHEAD * S_LEN);
    bf16* ao  = carve((size_t)S_LEN * DMODEL);
    (void)ws_size; (void)in_sizes; (void)n_in; (void)out_size;

    int nx = S_LEN * DMODEL, nwi = QKVN * DMODEL, nwo = DMODEL * DMODEL;
    cvt_f32_bf16<<<(nx / 8 + 255) / 256, 256, 0, stream>>>(x, xb, nx);
    cvt_f32_bf16<<<(nwi / 8 + 255) / 256, 256, 0, stream>>>(w_in, wib, nwi);
    cvt_f32_bf16<<<(nwo / 8 + 255) / 256, 256, 0, stream>>>(w_out, wob, nwo);

    qkv_gemm<<<dim3(QKVN / 128, S_LEN / 64), 256, 0, stream>>>(
        xb, wib, b_in, qh, kh, vt);

    attention<<<dim3(S_LEN / 256, NHEADS), 256, 0, stream>>>(
        qh, kh, vt, ao, flag);

    out_proj<<<dim3(DMODEL / 128, S_LEN / 64), 256, 0, stream>>>(
        ao, wob, b_out, (float*)d_out);
}
